// homo_meta_path_layer_2680059592915
// MI455X (gfx1250) — compile-verified, matrix operand order checked by reading, not run on hardware
//
#include <hip/hip_runtime.h>
#include <math.h>

#define NN 3072
#define DD 256
#define CATD 768
#define TAUV 0.8f
#define LAMV 0.5f
#define EPSV 1e-8f
#define SLOPEV 0.05f
#define MASKNEG (-3.0e4f)

typedef __attribute__((ext_vector_type(16))) _Float16 v16h;
typedef __attribute__((ext_vector_type(2)))  _Float16 v2h;
typedef __attribute__((ext_vector_type(8)))  float    v8f;
typedef __attribute__((ext_vector_type(2)))  float    f32x2;
typedef __attribute__((ext_vector_type(4)))  unsigned int u32x4;

union HVec {
    v16h h16;
    _Float16 h[16];
    u32x4 q[2];
};
union H8 {
    u32x4 q;
    _Float16 h[8];
};
union FAcc {
    v8f v;
    float f[8];
};

static __device__ __forceinline__ v8f wmma16(const HVec& a, const HVec& b, v8f c) {
    return __builtin_amdgcn_wmma_f32_16x16x32_f16(false, a.h16, false, b.h16,
                                                  (short)0, c, false, false);
}

// async copy of one 16B chunk global->LDS, tracked by ASYNCcnt
static __device__ __forceinline__ void async_b128(const _Float16* g, const _Float16* l) {
    unsigned loff = (unsigned)(size_t)(const void*)l;               // LDS byte offset
    unsigned long long ga = (unsigned long long)(size_t)(const void*)g;
    asm volatile("global_load_async_to_lds_b128 %0, %1, off"
                 :: "v"(loff), "v"(ga) : "memory");
}
static __device__ __forceinline__ void wait_async0() {
    asm volatile("s_wait_asynccnt 0x0" ::: "memory");
}

// ---------------------------------------------------------------------------
// C[NN,DD] = act(alpha * A @ B16^T + beta*Cin + bias)
// A:[NN,K] f32, B16:[DD,K] f16 (pre-converted weights).
// Tile BM=64, BN=128, BK=32. 256 threads = 8 waves (2x4 wave grid, 32x32 each).
// B tile staged via GLOBAL_LOAD_ASYNC_TO_LDS_B128; A tile via VALU f32->f16.
// act: 0=none, 1=tanh, 2=elu
// ---------------------------------------------------------------------------
__global__ __launch_bounds__(256)
void gemm_bt_wmma(const float* __restrict__ A, const _Float16* __restrict__ B16,
                  const float* __restrict__ Cin, const float* __restrict__ bias,
                  float* __restrict__ C, int K, float alpha, float beta, int act)
{
    const int LDT = 40; // halves per LDS row (80B, 16B aligned)
    __shared__ __align__(16) _Float16 As[64 * 40];
    __shared__ __align__(16) _Float16 Bs[128 * 40];

    const int t    = threadIdx.x;
    const int lane = t & 31;
    const int wid  = t >> 5;
    const int m0   = blockIdx.y * 64;
    const int n0   = blockIdx.x * 128;
    const int mb   = (wid & 1) * 32;
    const int nb   = (wid >> 1) * 32;
    const int lm   = lane & 15;
    const int kb   = (lane >> 4) ? 8 : 0;

    FAcc acc[2][2];
    for (int a0 = 0; a0 < 2; ++a0)
        for (int b0 = 0; b0 < 2; ++b0)
            for (int e = 0; e < 8; ++e) acc[a0][b0].f[e] = 0.0f;

    for (int k0 = 0; k0 < K; k0 += 32) {
        // stage B tile: 128x32 halves = 512 x 16B, async global->LDS
        for (int it = 0; it < 2; ++it) {
            int idx = t + it * 256;
            int row = idx >> 2, q = idx & 3;
            async_b128(B16 + (size_t)(n0 + row) * K + k0 + q * 8,
                       &Bs[row * LDT + q * 8]);
        }
        // stage A tile: 64x32 f32 -> f16 (1024 pairs)
        for (int it = 0; it < 4; ++it) {
            int idx = t + it * 256;
            int m = idx >> 4, kp = idx & 15;
            f32x2 f = *(const f32x2*)(A + (size_t)(m0 + m) * K + k0 + 2 * kp);
            v2h hv; hv[0] = (_Float16)f.x; hv[1] = (_Float16)f.y;
            *(v2h*)&As[m * LDT + 2 * kp] = hv;
        }
        wait_async0();
        __syncthreads();

        HVec af[2], bf[2];
        for (int sm = 0; sm < 2; ++sm) {
            const _Float16* p = &As[(mb + sm * 16 + lm) * LDT + kb];
            af[sm].q[0] = *(const u32x4*)p;
            af[sm].q[1] = *(const u32x4*)(p + 16);
        }
        for (int sn = 0; sn < 2; ++sn) {
            const _Float16* p = &Bs[(nb + sn * 16 + lm) * LDT + kb];
            bf[sn].q[0] = *(const u32x4*)p;
            bf[sn].q[1] = *(const u32x4*)(p + 16);
        }
        for (int sm = 0; sm < 2; ++sm)
            for (int sn = 0; sn < 2; ++sn)
                acc[sm][sn].v = wmma16(af[sm], bf[sn], acc[sm][sn].v);
        __syncthreads();
    }

    const int mhi = (lane >> 4) ? 8 : 0;
    for (int sm = 0; sm < 2; ++sm)
        for (int sn = 0; sn < 2; ++sn) {
            int gn = n0 + nb + sn * 16 + lm;
            float bv = bias ? bias[gn] : 0.0f;
            for (int e = 0; e < 8; ++e) {
                int gm = m0 + mb + sm * 16 + e + mhi;
                float r = acc[sm][sn].f[e] * alpha;
                if (beta != 0.0f) r += beta * Cin[(size_t)gm * DD + gn];
                r += bv;
                if (act == 1)       r = tanhf(r);
                else if (act == 2)  r = (r > 0.0f) ? r : (expf(r) - 1.0f);
                C[(size_t)gm * DD + gn] = r;
            }
        }
}

// ---------------------------------------------------------------------------
// Fused masked-softmax attention:  out = softmax(mask(leakyrelu(s_i+d_j))) @ feat
// One block = 16 output rows. Scores cached as f16 in LDS (16 x 3072 strip).
// featT: [DD][NN] f16 (pre-transposed) so B fragments are contiguous b128 loads.
// ---------------------------------------------------------------------------
__global__ __launch_bounds__(256)
void attn_kernel(const float* __restrict__ adj, const float* __restrict__ s,
                 const float* __restrict__ d, const _Float16* __restrict__ featT,
                 float* __restrict__ out)
{
    const int LDR = NN + 8;
    __shared__ __align__(16) _Float16 sc[16 * (NN + 8)];
    __shared__ float red[16 * 16];
    __shared__ float rowstat[32]; // [0..15]=rowmax, [16..31]=1/rowsum

    const int t  = threadIdx.x;
    const int r  = t >> 4;   // row 0..15
    const int jg = t & 15;   // 16 threads sweep each row
    const int i0 = blockIdx.x * 16;
    const int i  = i0 + r;

    // phase 1: scores + row max
    const float sI = s[i];
    float lmax = MASKNEG;
    for (int j = jg; j < NN; j += 16) {
        float a = adj[(size_t)i * NN + j];
        float v = sI + d[j];
        v = (v > 0.0f) ? v : SLOPEV * v;
        v = (a > 0.0f) ? v : MASKNEG;
        sc[r * LDR + j] = (_Float16)v;
        lmax = fmaxf(lmax, v);
        __builtin_prefetch(adj + (size_t)i * NN + j + 128, 0, 1);
    }
    red[r * 16 + jg] = lmax;
    __syncthreads();
    if (jg == 0) {
        float m = red[r * 16];
        for (int q = 1; q < 16; ++q) m = fmaxf(m, red[r * 16 + q]);
        rowstat[r] = m;
    }
    __syncthreads();

    // phase 2: sum of exp
    const float rm = rowstat[r];
    float lsum = 0.0f;
    for (int j = jg; j < NN; j += 16)
        lsum += expf((float)sc[r * LDR + j] - rm);
    red[r * 16 + jg] = lsum;
    __syncthreads();
    if (jg == 0) {
        float sm = 0.0f;
        for (int q = 0; q < 16; ++q) sm += red[r * 16 + q];
        rowstat[16 + r] = 1.0f / sm;
    }
    __syncthreads();

    // phase 3: P @ feat via WMMA (each wave owns 32 output feature cols)
    const int lane = t & 31;
    const int wid  = t >> 5;
    const int lm   = lane & 15;
    const int kb   = (lane >> 4) ? 8 : 0;
    const int nbase = wid * 32;
    const float rmax = rowstat[lm];
    const float rinv = rowstat[16 + lm];

    FAcc acc[2];
    for (int sn = 0; sn < 2; ++sn)
        for (int e = 0; e < 8; ++e) acc[sn].f[e] = 0.0f;

    for (int j0 = 0; j0 < NN; j0 += 32) {
        HVec af;
        {
            const _Float16* p = &sc[lm * LDR + j0 + kb];
            H8 lo, hi;
            lo.q = *(const u32x4*)p;
            hi.q = *(const u32x4*)(p + 16);
            for (int e = 0; e < 8; ++e) {
                af.h[e]     = (_Float16)(expf((float)lo.h[e] - rmax) * rinv);
                af.h[8 + e] = (_Float16)(expf((float)hi.h[e] - rmax) * rinv);
            }
        }
        for (int sn = 0; sn < 2; ++sn) {
            int n = nbase + sn * 16 + lm;
            const _Float16* p = featT + (size_t)n * NN + j0 + kb;
            HVec bf;
            bf.q[0] = *(const u32x4*)p;
            bf.q[1] = *(const u32x4*)(p + 16);
            acc[sn].v = wmma16(af, bf, acc[sn].v);
        }
    }

    const int mhi = (lane >> 4) ? 8 : 0;
    for (int sn = 0; sn < 2; ++sn) {
        int gn = nbase + sn * 16 + lm;
        for (int e = 0; e < 8; ++e) {
            int gm = i0 + e + mhi;
            out[(size_t)gm * DD + gn] = acc[sn].f[e];
        }
    }
}

// ---------------------------------------------------------------------------
// Contrast tiles: num = p1 @ p2^T (64x64 tile, WMMA), S=exp(num/(n1*n2*tau)),
// accumulate row/col sums via LDS + global atomics, capture diag.
// ---------------------------------------------------------------------------
__global__ __launch_bounds__(256)
void contrast_tile(const float* __restrict__ p1, const float* __restrict__ p2,
                   const float* __restrict__ n1, const float* __restrict__ n2,
                   float* __restrict__ rowsum, float* __restrict__ colsum,
                   float* __restrict__ diag)
{
    const int LDT = 40;
    __shared__ __align__(16) _Float16 As[64 * 40];
    __shared__ __align__(16) _Float16 Bs[64 * 40];
    __shared__ float rs[64], cs[64];

    const int t    = threadIdx.x;
    const int lane = t & 31;
    const int wid  = t >> 5;
    const int m0   = blockIdx.y * 64;
    const int n0   = blockIdx.x * 64;
    const int mb   = (wid & 3) * 16;
    const int nb   = (wid >> 2) * 32;
    const int lm   = lane & 15;
    const int kb   = (lane >> 4) ? 8 : 0;

    if (t < 64) { rs[t] = 0.0f; cs[t] = 0.0f; }

    FAcc acc[2];
    for (int sn = 0; sn < 2; ++sn)
        for (int e = 0; e < 8; ++e) acc[sn].f[e] = 0.0f;

    for (int k0 = 0; k0 < DD; k0 += 32) {
        for (int it = 0; it < 4; ++it) {
            int idx = t + it * 256;
            int m = idx >> 4, kp = idx & 15;
            f32x2 fa = *(const f32x2*)(p1 + (size_t)(m0 + m) * DD + k0 + 2 * kp);
            f32x2 fb = *(const f32x2*)(p2 + (size_t)(n0 + m) * DD + k0 + 2 * kp);
            v2h ha; ha[0] = (_Float16)fa.x; ha[1] = (_Float16)fa.y;
            v2h hb; hb[0] = (_Float16)fb.x; hb[1] = (_Float16)fb.y;
            *(v2h*)&As[m * LDT + 2 * kp] = ha;
            *(v2h*)&Bs[m * LDT + 2 * kp] = hb;
        }
        __syncthreads();
        HVec af, bf;
        const _Float16* pa = &As[(mb + lm) * LDT + kb];
        af.q[0] = *(const u32x4*)pa;
        af.q[1] = *(const u32x4*)(pa + 16);
        for (int sn = 0; sn < 2; ++sn) {
            const _Float16* pb = &Bs[(nb + sn * 16 + lm) * LDT + kb];
            bf.q[0] = *(const u32x4*)pb;
            bf.q[1] = *(const u32x4*)(pb + 16);
            acc[sn].v = wmma16(af, bf, acc[sn].v);
        }
        __syncthreads();
    }

    const int mhi = (lane >> 4) ? 8 : 0;
    for (int sn = 0; sn < 2; ++sn) {
        int ln = nb + sn * 16 + lm;
        int gn = n0 + ln;
        float nj = n2[gn];
        for (int e = 0; e < 8; ++e) {
            int lmm = mb + e + mhi;
            int gm = m0 + lmm;
            float ni = n1[gm];
            float sv = expf(acc[sn].f[e] / (ni * nj * TAUV));
            atomicAdd(&rs[lmm], sv);
            atomicAdd(&cs[ln], sv);
            if (gm == gn) diag[gm] = sv;
        }
    }
    __syncthreads();
    if (t < 64) {
        atomicAdd(rowsum + m0 + t, rs[t]);
        atomicAdd(colsum + n0 + t, cs[t]);
    }
}

// ---------------------------------------------------------------------------
// small helper kernels
// ---------------------------------------------------------------------------
__global__ void sd_kernel(const float* __restrict__ x, const float* __restrict__ asrc,
                          const float* __restrict__ adst, float* __restrict__ s,
                          float* __restrict__ d)
{
    int i = blockIdx.x * blockDim.x + threadIdx.x;
    if (i >= NN) return;
    float a = 0.0f, b = 0.0f;
    for (int k = 0; k < DD; ++k) {
        float v = x[(size_t)i * DD + k];
        a += v * asrc[k];
        b += v * adst[k];
    }
    s[i] = a; d[i] = b;
}

__global__ void rownorm_kernel(const float* __restrict__ p, float* __restrict__ nrm)
{
    int i = blockIdx.x * blockDim.x + threadIdx.x;
    if (i >= NN) return;
    float a = 0.0f;
    for (int k = 0; k < DD; ++k) {
        float v = p[(size_t)i * DD + k];
        a += v * v;
    }
    nrm[i] = sqrtf(a);
}

__global__ void convT16_kernel(const float* __restrict__ in, _Float16* __restrict__ outT)
{
    int idx = blockIdx.x * blockDim.x + threadIdx.x; // over NN*DD (output order)
    int dd = idx / NN;
    int i  = idx - dd * NN;
    outT[idx] = (_Float16)in[(size_t)i * DD + dd];
}

__global__ void conv16_kernel(const float* __restrict__ in, _Float16* __restrict__ out, int n)
{
    int i = blockIdx.x * blockDim.x + threadIdx.x;
    if (i < n) out[i] = (_Float16)in[i];
}

__global__ void concat_kernel(const float* __restrict__ F, const float* __restrict__ h1,
                              const float* __restrict__ h2, float* __restrict__ cat)
{
    int idx = blockIdx.x * blockDim.x + threadIdx.x; // over NN*CATD
    int i = idx / CATD;
    int c = idx - i * CATD;
    float v;
    if (c < 256)      v = F [(size_t)i * DD + c];
    else if (c < 512) v = h1[(size_t)i * DD + c - 256];
    else              v = h2[(size_t)i * DD + c - 512];
    cat[idx] = v;
}

__global__ void zero_kernel(float* __restrict__ p, int n)
{
    int i = blockIdx.x * blockDim.x + threadIdx.x;
    if (i < n) p[i] = 0.0f;
}

__global__ __launch_bounds__(256)
void loss_reduce(const float* __restrict__ diag, const float* __restrict__ rowsum,
                 const float* __restrict__ colsum, float* __restrict__ out, float w)
{
    __shared__ float s1[256], s2[256];
    int t = threadIdx.x;
    float a = 0.0f, b = 0.0f;
    for (int i = t; i < NN; i += 256) {
        float dg = diag[i];
        a += -logf(dg / (rowsum[i] + EPSV));
        b += -logf(dg / (colsum[i] + EPSV));
    }
    s1[t] = a; s2[t] = b;
    __syncthreads();
    for (int st = 128; st > 0; st >>= 1) {
        if (t < st) { s1[t] += s1[t + st]; s2[t] += s2[t + st]; }
        __syncthreads();
    }
    if (t == 0)
        atomicAdd(out, w * (LAMV * s1[0] + (1.0f - LAMV) * s2[0]) / (float)NN);
}

// ---------------------------------------------------------------------------
extern "C" void kernel_launch(void* const* d_in, const int* in_sizes, int n_in,
                              void* d_out, int out_size, void* d_ws, size_t ws_size,
                              hipStream_t stream)
{
    (void)in_sizes; (void)n_in; (void)out_size; (void)ws_size;

    const float* adj[3] = {(const float*)d_in[0], (const float*)d_in[1], (const float*)d_in[2]};
    const float* F        = (const float*)d_in[3];
    const float* gat_W    = (const float*)d_in[5];
    const float* gat_asrc = (const float*)d_in[6];
    const float* gat_adst = (const float*)d_in[7];
    const float* res_W    = (const float*)d_in[8];
    const float* mlp_W    = (const float*)d_in[9];
    const float* W1       = (const float*)d_in[10];
    const float* b1       = (const float*)d_in[11];
    const float* W2       = (const float*)d_in[12];
    const float* b2       = (const float*)d_in[13];

    float* out  = (float*)d_out;
    float* loss = out + (size_t)NN * DD;

    // workspace carving (all chunks multiples of 16B)
    float* ws = (float*)d_ws;
    size_t off = 0;
    auto take = [&](size_t n) { float* p = ws + off; off += n; return p; };
    _Float16* featT = (_Float16*)take((size_t)NN * DD / 2);
    _Float16* gW16  = (_Float16*)take(2 * DD * DD / 2);   // both GAT layers
    _Float16* rW16  = (_Float16*)take(2 * DD * DD / 2);   // both residual layers
    _Float16* mW16  = (_Float16*)take((size_t)DD * CATD / 2);
    _Float16* W116  = (_Float16*)take(DD * DD / 2);
    _Float16* W216  = (_Float16*)take(DD * DD / 2);
    float* xbuf = take((size_t)NN * DD);
    float* h1[3], * h2[3];
    for (int k = 0; k < 3; ++k) h1[k] = take((size_t)NN * DD);
    for (int k = 0; k < 3; ++k) h2[k] = take((size_t)NN * DD);
    float* cat  = take((size_t)NN * CATD);
    float* res1 = take((size_t)NN * DD);
    float* res2 = take((size_t)NN * DD);
    float* ptmp = take((size_t)NN * DD);
    float* p1   = take((size_t)NN * DD);
    float* p2   = take((size_t)NN * DD);
    float* sbuf = take(NN);
    float* dbuf = take(NN);
    float* nn1  = take(NN);
    float* nn2  = take(NN);
    float* rsum = take(NN);
    float* csum = take(NN);
    float* dgbf = take(NN);

    auto gemm = [&](const float* A, const _Float16* B16, const float* Cin,
                    const float* bias, float* C, int K, float alpha, float beta, int act) {
        dim3 grid(DD / 128, NN / 64);
        gemm_bt_wmma<<<grid, 256, 0, stream>>>(A, B16, Cin, bias, C, K, alpha, beta, act);
    };

    // ------------------- one-time weight conversions (f32 -> f16) -------------------
    conv16_kernel<<<(2 * DD * DD) / 256, 256, 0, stream>>>(gat_W, gW16, 2 * DD * DD);
    conv16_kernel<<<(2 * DD * DD) / 256, 256, 0, stream>>>(res_W, rW16, 2 * DD * DD);
    conv16_kernel<<<(DD * CATD) / 256, 256, 0, stream>>>(mlp_W, mW16, DD * CATD);
    conv16_kernel<<<(DD * DD) / 256, 256, 0, stream>>>(W1, W116, DD * DD);
    conv16_kernel<<<(DD * DD) / 256, 256, 0, stream>>>(W2, W216, DD * DD);

    // ------------------- layer 0 (h[0][k] == F for all k) -------------------
    gemm(F, gW16, nullptr, nullptr, xbuf, DD, 1.0f, 0.0f, 1);                // x = tanh(F@W0^T)
    sd_kernel<<<NN / 256, 256, 0, stream>>>(xbuf, gat_asrc, gat_adst, sbuf, dbuf);
    convT16_kernel<<<(NN * DD) / 256, 256, 0, stream>>>(F, featT);
    for (int k = 0; k < 3; ++k)
        attn_kernel<<<NN / 16, 256, 0, stream>>>(adj[k], sbuf, dbuf, featT, h1[k]);
    gemm(F, rW16, h1[0], nullptr, h1[0], DD, 0.2f, 1.0f, 0);                 // x0 += 0.2*F@R0^T
    gemm(F, rW16, h1[1], nullptr, h1[1], DD, 0.2f, 1.0f, 0);                 // x1 += 0.2*F@R0^T
    gemm(F, rW16, h1[1], nullptr, h1[1], DD, 0.2f, 1.0f, 0);                 // x1 += 0.2*F@R0^T (pro2)

    // ------------------- layer 1 -------------------
    const _Float16* gW16_1 = gW16 + DD * DD;
    const _Float16* rW16_1 = rW16 + DD * DD;
    for (int k = 0; k < 3; ++k) {
        gemm(h1[k], gW16_1, nullptr, nullptr, xbuf, DD, 1.0f, 0.0f, 1);
        sd_kernel<<<NN / 256, 256, 0, stream>>>(xbuf, gat_asrc + DD, gat_adst + DD, sbuf, dbuf);
        convT16_kernel<<<(NN * DD) / 256, 256, 0, stream>>>(h1[k], featT);
        attn_kernel<<<NN / 16, 256, 0, stream>>>(adj[k], sbuf, dbuf, featT, h2[k]);
    }
    gemm(h1[0], rW16_1, h2[0], nullptr, h2[0], DD, 0.2f, 1.0f, 0);
    gemm(h1[1], rW16_1, h2[1], nullptr, h2[1], DD, 0.2f, 1.0f, 0);
    gemm(h1[2], rW16_1, h2[1], nullptr, h2[1], DD, 0.2f, 1.0f, 0);

    // ------------------- MLP over concat -------------------
    float* results[3] = {out, res1, res2};
    for (int k = 0; k < 3; ++k) {
        concat_kernel<<<(NN * CATD) / 256, 256, 0, stream>>>(F, h1[k], h2[k], cat);
        gemm(cat, mW16, nullptr, nullptr, results[k], CATD, 1.0f, 0.0f, 2);
    }

    // ------------------- contrastive losses -------------------
    gemm(out, W116, nullptr, b1, ptmp, DD, 1.0f, 0.0f, 2);                   // p1 = proj(result0)
    gemm(ptmp, W216, nullptr, b2, p1, DD, 1.0f, 0.0f, 0);
    rownorm_kernel<<<NN / 256, 256, 0, stream>>>(p1, nn1);
    zero_kernel<<<1, 32, 0, stream>>>(loss, 1);

    for (int c = 0; c < 2; ++c) {
        const float* zk = (c == 0) ? res1 : res2;
        gemm(zk, W116, nullptr, b1, ptmp, DD, 1.0f, 0.0f, 2);                // p2 = proj(result[c+1])
        gemm(ptmp, W216, nullptr, b2, p2, DD, 1.0f, 0.0f, 0);
        rownorm_kernel<<<NN / 256, 256, 0, stream>>>(p2, nn2);
        zero_kernel<<<(3 * NN) / 256, 256, 0, stream>>>(rsum, 3 * NN);       // rsum,csum,dg contiguous
        contrast_tile<<<dim3(NN / 64, NN / 64), 256, 0, stream>>>(p1, p2, nn1, nn2,
                                                                  rsum, csum, dgbf);
        loss_reduce<<<1, 256, 0, stream>>>(dgbf, rsum, csum, loss, 0.5f);
    }
}
